// ArithmeticResidualBlock_28209345200789
// MI455X (gfx1250) — compile-verified, matrix operand order checked by reading, not run on hardware
//
#include <hip/hip_runtime.h>
#include <hip/hip_bf16.h>
#include <cstdint>

typedef __attribute__((ext_vector_type(16))) _Float16 v16h;
typedef __attribute__((ext_vector_type(8)))  _Float16 v8h;
typedef __attribute__((ext_vector_type(8)))  float    v8f;
typedef __attribute__((ext_vector_type(4)))  unsigned int u32x4;
typedef __attribute__((ext_vector_type(8)))  int      i32x8;
typedef __attribute__((ext_vector_type(4)))  int      i32x4;

#define K_DIG    10
#define NPRIME   9
#define D_MODEL  768
#define MH_DIM   128
#define NLOGIT   200
#define SEQ_LEN  2048
#define NTOK     32768          // 16 * 2048
#define TILE_M   16
#define NBLK     2048           // NTOK / TILE_M
#define HP_OFF   25165824LL     // 16*2048*768
#define DA_SZ    327680LL       // 16*2048*10
#define HS_STRIDE 780           // 768 + 3*4 pad DWORDs (TDM pad: 4 dwords per 256)

static constexpr long long M_TOTAL_LL = 247357937827LL;
static constexpr float TWO_PI_F = 6.28318530717958647692f;

// ---- compile-time tables: 10^k mod p (as float) and CRT coefficients ----
struct Tables {
  float     pow10f[NPRIME][K_DIG];
  long long crt[NPRIME];
  long long prime[NPRIME];
  constexpr Tables() : pow10f{}, crt{}, prime{} {
    const long long P[NPRIME] = {7,11,13,17,19,23,29,31,37};
    long long M = 1;
    for (int i = 0; i < NPRIME; ++i) { prime[i] = P[i]; M *= P[i]; }
    for (int i = 0; i < NPRIME; ++i) {
      long long v = 1 % P[i];
      for (int k = 0; k < K_DIG; ++k) { pow10f[i][k] = (float)v; v = (v * 10) % P[i]; }
      long long Mi  = M / P[i];
      long long mi  = Mi % P[i];
      long long inv = 1;
      for (long long x = 1; x < P[i]; ++x) if ((mi * x) % P[i] == 1) { inv = x; break; }
      crt[i] = (Mi % M) * inv % M;
    }
  }
};
static constexpr Tables TBL{};

// TDM pads 4 DWORDs after every 256 stored DWORDs: col -> col + 4*(col/256)
__device__ __forceinline__ int lcol(int c) { return c + ((c >> 8) << 2); }

// ---------------- WMMA fragment helpers ----------------
// 16-bit A 16x32 layout: lane (g=l>>4, m=l&15); element e<8 -> K = kk+g*8+e,
// e>=8 -> K = kk+16+g*8+(e-8). B 32x16 mirrors with N=l&15.
__device__ __forceinline__ v16h load_a_from_f32(const float* rowp) {
  const float4* p0 = (const float4*)rowp;
  const float4* p1 = (const float4*)(rowp + 16);
  float4 f0 = p0[0], f1 = p0[1], f2 = p1[0], f3 = p1[1];
  v16h a;
  a[0]=(_Float16)f0.x; a[1]=(_Float16)f0.y; a[2]=(_Float16)f0.z; a[3]=(_Float16)f0.w;
  a[4]=(_Float16)f1.x; a[5]=(_Float16)f1.y; a[6]=(_Float16)f1.z; a[7]=(_Float16)f1.w;
  a[8]=(_Float16)f2.x; a[9]=(_Float16)f2.y; a[10]=(_Float16)f2.z; a[11]=(_Float16)f2.w;
  a[12]=(_Float16)f3.x; a[13]=(_Float16)f3.y; a[14]=(_Float16)f3.z; a[15]=(_Float16)f3.w;
  return a;
}
__device__ __forceinline__ v16h load_frag_f16(const _Float16* p) {
  v8h lo = *(const v8h*)p;
  v8h hi = *(const v8h*)(p + 16);
  return __builtin_shufflevector(lo, hi, 0,1,2,3,4,5,6,7,8,9,10,11,12,13,14,15);
}
__device__ __forceinline__ v8f wmma_f16(v16h a, v16h b, v8f c) {
  return __builtin_amdgcn_wmma_f32_16x16x32_f16(false, a, false, b, (short)0, c, false, false);
}

// ---------------- modular helpers ----------------
__device__ __forceinline__ long long decq(float y, float x, float pf, long long pl) {
  float th = atan2f(y, x);
  long long r = (long long)roundf(th * (pf / TWO_PI_F));
  r %= pl; if (r < 0) r += pl;
  return r;
}
__device__ __forceinline__ long long modpow6(long long base, long long e, long long m) {
  long long res = 1, b = base % m;
#pragma unroll
  for (int i = 0; i < 6; ++i) {
    if (e & 1) res = res * b % m;
    b = b * b % m;
    e >>= 1;
  }
  return res;
}

// ---------------- prep kernels ----------------
__global__ void prep_weights(const float* __restrict__ W1, const float* __restrict__ W2,
                             _Float16* __restrict__ W1T, _Float16* __restrict__ W2T) {
  int i = blockIdx.x * blockDim.x + threadIdx.x;
  if (i < MH_DIM * D_MODEL) {                 // W1T[n][k] = W1[k][n], n<128, k<768
    int n = i / D_MODEL, k = i % D_MODEL;
    W1T[i] = (_Float16)W1[k * MH_DIM + n];
  }
  if (i < 208 * MH_DIM) {                     // W2T[n][k] = W2[k][n], padded to 208 rows
    int n = i / MH_DIM, k = i % MH_DIM;
    W2T[i] = (_Float16)((n < NLOGIT) ? W2[k * NLOGIT + n] : 0.f);
  }
}

__global__ void prep_meta(const int* __restrict__ ids, int* __restrict__ meta) {
  __shared__ int s_min, s_eq;
  if (threadIdx.x == 0) { s_min = 1 << 30; s_eq = 0; }
  __syncthreads();
  int b = blockIdx.x;
  int lmin = 1 << 30, leq = 0;
  for (int i = threadIdx.x; i < SEQ_LEN; i += blockDim.x) {
    int id = ids[b * SEQ_LEN + i];
    if (id >= 20 && id <= 24) lmin = (i < lmin) ? i : lmin;
    if (id == 28) leq = 1;
  }
  atomicMin(&s_min, lmin);
  if (leq) atomicOr(&s_eq, 1);
  __syncthreads();
  if (threadIdx.x == 0) {
    int opi = -1;
    if (s_min < SEQ_LEN) opi = ids[b * SEQ_LEN + s_min] - 20;
    meta[2 * b]     = opi;
    meta[2 * b + 1] = s_eq;
  }
}

// ---------------- fused main kernel: 2048 blocks x 128 threads (4 waves) ----------------
__global__ void __launch_bounds__(128)
fused_tile(const float* __restrict__ h, const _Float16* __restrict__ W1T,
           const float* __restrict__ b1, const _Float16* __restrict__ W2T,
           const float* __restrict__ b2, const float* __restrict__ Winj,
           const float* __restrict__ binj, const float* __restrict__ gate,
           const int* __restrict__ meta, float* __restrict__ out) {
  __shared__ __align__(16) float    hS[TILE_M * HS_STRIDE]; // f32 h tile, TDM-padded rows
  __shared__ __align__(16) _Float16 logitsS[TILE_M][NLOGIT];
  __shared__ __align__(16) char     uS[4352];               // phase-overlaid union
  _Float16 (*hidS)[136]          = (_Float16 (*)[136])uS;                 // GEMM1->GEMM2
  float (*digS)[20]              = (float (*)[20])uS;                     // post phase
  unsigned long long (*sumS)[5]  = (unsigned long long (*)[5])(uS + 1280);
  float (*msbS)[12]              = (float (*)[12])(uS + 1920);

  const int tid  = threadIdx.x;
  const int w    = tid >> 5;          // wave id (wave32)
  const int l    = tid & 31;          // lane
  const int ml   = l & 15;
  const int g    = l >> 4;
  const int row0 = blockIdx.x * TILE_M;
  const int brow = row0 >> 11;        // batch row (2048 tokens per row)

  // ---- stage 0: TDM DMA: 16x768 f32 tile, global -> LDS (padded 4dw/256dw) ----
  if (w == 0) {
    unsigned ldsaddr = (unsigned)(uintptr_t)(void*)&hS[0];  // flat low 32b == LDS addr
    unsigned long long ga =
        (unsigned long long)(uintptr_t)(h + (size_t)row0 * D_MODEL);
    u32x4 g0;
    g0[0] = 1u;                                             // count=1, user mode
    g0[1] = ldsaddr;                                        // lds_addr
    g0[2] = (unsigned)(ga & 0xffffffffu);                   // global_addr[31:0]
    g0[3] = (unsigned)((ga >> 32) & 0x01ffffffu)            // global_addr[56:32]
          | (2u << 30);                                     // type = 2 (image)
    i32x8 g1;
    g1[0] = (int)((2u << 16)      // data_size = 4B
                | (1u << 20)      // pad_enable
                | (7u << 22)      // pad_interval: 256 DWORDs
                | (3u << 25));    // pad_amount: 4 DWORDs
    g1[1] = (int)(768u << 16);    // abar_addr=0 | tensor_dim0[15:0]=768
    g1[2] = 0;                    // tensor_dim0[31:16]=0 | tensor_dim1[15:0]=0
    g1[3] = (int)(16u             // tensor_dim1[31:16] = 16 (dim1 = 1<<20, no OOB)
                | (768u << 16));  // tile_dim0 = 768
    g1[4] = 16;                   // tile_dim1 = 16, tile_dim2 = 0 (2-D)
    g1[5] = 768;                  // tensor_dim0_stride[31:0] = 768
    g1[6] = 0;                    // stride hi | dim1_stride lo (unused for 2-D)
    g1[7] = 0;
    i32x4 g2 = {0, 0, 0, 0};
    i32x4 g3 = {0, 0, 0, 0};
    i32x8 g4 = {0, 0, 0, 0, 0, 0, 0, 0};   // 6-arg form: extra descriptor group (zeros)
    __builtin_amdgcn_tensor_load_to_lds(g0, g1, g2, g3, g4, 0);
    __builtin_amdgcn_s_wait_tensorcnt(0);
  }
  __syncthreads();

  // ---- stage 1: GEMM1 hid = relu(h @ W1 + b1), M=16 N=128 K=768 ----
  {
    v8f acc0 = {}, acc1 = {};
    const int nt0 = w, nt1 = w + 4;
    const _Float16* bp0 = W1T + (size_t)(nt0 * 16 + ml) * D_MODEL + g * 8;
    const _Float16* bp1 = W1T + (size_t)(nt1 * 16 + ml) * D_MODEL + g * 8;
    const float*    ap  = hS + ml * HS_STRIDE;
    __builtin_prefetch(bp0, 0, 3);
    __builtin_prefetch(bp1, 0, 3);
#pragma unroll 4
    for (int kk = 0; kk < D_MODEL; kk += 32) {
      v16h a   = load_a_from_f32(ap + lcol(kk + g * 8));
      v16h bb0 = load_frag_f16(bp0 + kk);
      v16h bb1 = load_frag_f16(bp1 + kk);
      acc0 = wmma_f16(a, bb0, acc0);
      acc1 = wmma_f16(a, bb1, acc1);
    }
    float bv0 = b1[nt0 * 16 + ml];
    float bv1 = b1[nt1 * 16 + ml];
#pragma unroll
    for (int j = 0; j < 8; ++j) {
      int m = j + 8 * g;
      hidS[m][nt0 * 16 + ml] = (_Float16)fmaxf(acc0[j] + bv0, 0.f);
      hidS[m][nt1 * 16 + ml] = (_Float16)fmaxf(acc1[j] + bv1, 0.f);
    }
  }
  __syncthreads();

  // ---- stage 2: GEMM2 logits = hid @ W2 + b2, M=16 N=200(pad 208) K=128 ----
  for (int nt = w; nt < 13; nt += 4) {
    v8f acc = {};
    const _Float16* bp = W2T + (size_t)(nt * 16 + ml) * MH_DIM + g * 8;
    const _Float16* ap = &hidS[ml][g * 8];
    __builtin_prefetch(bp, 0, 3);
#pragma unroll
    for (int kk = 0; kk < MH_DIM; kk += 32) {
      v16h a = load_frag_f16(ap + kk);
      v16h b = load_frag_f16(bp + kk);
      acc = wmma_f16(a, b, acc);
    }
    int n = nt * 16 + ml;
    if (n < NLOGIT) {
      float bv = b2[n];
#pragma unroll
      for (int j = 0; j < 8; ++j) logitsS[j + 8 * g][n] = (_Float16)(acc[j] + bv);
    }
  }
  __syncthreads();

  // ---- stage 3: softmax expected digits (320 groups across 128 threads) ----
  if (tid < 80) ((unsigned long long*)&sumS[0][0])[tid] = 0ull;  // zero CRT sums
  for (int grp = tid; grp < TILE_M * 20; grp += 128) {
    int m = grp / 20, ok = grp % 20;
    const _Float16* lp = &logitsS[m][ok * 10];
    float v[10];
#pragma unroll
    for (int c = 0; c < 10; ++c) v[c] = (float)lp[c];
    float mx = v[0];
#pragma unroll
    for (int c = 1; c < 10; ++c) mx = fmaxf(mx, v[c]);
    float s = 0.f, a = 0.f;
#pragma unroll
    for (int c = 0; c < 10; ++c) { float e = __expf(v[c] - mx); s += e; a += e * (float)c; }
    float dg = a / s;
    digS[m][ok] = dg;
    int o = ok / 10, k = ok % 10;
    size_t t = (size_t)(row0 + m);
    out[HP_OFF + (o ? DA_SZ : 0) + t * 10 + k] = dg;
  }
  __syncthreads();

  // ---- stage 4: per (token, prime) residue math, CRT partial sums ----
  if (tid < TILE_M * NPRIME) {
    int m = tid / NPRIME, i = tid % NPRIME;
    const long long pl = TBL.prime[i];
    const float     pf = (float)pl;
    float ta = 0.f, tb = 0.f;
#pragma unroll
    for (int k = 0; k < K_DIG; ++k) {
      ta += digS[m][k]      * TBL.pow10f[i][k];
      tb += digS[m][10 + k] * TBL.pow10f[i][k];
    }
    float tha = TWO_PI_F * ta / pf, thb = TWO_PI_F * tb / pf;
    float ca = cosf(tha), sa = sinf(tha), cb = cosf(thb), sb = sinf(thb);
    long long ra = decq(sa, ca, pf, pl);
    long long rb = decq(sb, cb, pf, pl);
    long long r_add = decq(sa * cb + ca * sb, ca * cb - sa * sb, pf, pl);
    long long r_sub = decq(sa * cb - ca * sb, ca * cb + sa * sb, pf, pl);
    long long r_mul = (ra * rb) % pl;
    long long r_exp = modpow6(ra, rb, pl);
    long long r_div = (rb == 0) ? 0 : (ra * modpow6(rb, pl - 2, pl)) % pl;
    unsigned long long cc = (unsigned long long)TBL.crt[i];
    atomicAdd(&sumS[m][0], (unsigned long long)r_add * cc);
    atomicAdd(&sumS[m][1], (unsigned long long)r_sub * cc);
    atomicAdd(&sumS[m][2], (unsigned long long)r_mul * cc);
    atomicAdd(&sumS[m][3], (unsigned long long)r_exp * cc);
    atomicAdd(&sumS[m][4], (unsigned long long)r_div * cc);
  }
  __syncthreads();

  // ---- stage 5: finalize digits, select op, MSB reorder ----
  if (tid < TILE_M) {
    int m = tid;
    const long long M = M_TOTAL_LL;
    long long n[5];
#pragma unroll
    for (int j = 0; j < 5; ++j) n[j] = (long long)(sumS[m][j] % (unsigned long long)M);
    int opi = meta[2 * brow];
    float valid = (opi >= 0) ? 1.f : 0.f;
    int gi = opi < 0 ? 0 : (opi > 4 ? 4 : opi);
    long long nsel; float sgn = 0.f;
    if (gi == 1) {
      long long ns = n[1];
      if (ns > M / 2) ns -= M;
      if (ns < 0) { sgn = 1.f; ns = -ns; }
      nsel = ns;
    } else {
      nsel = (gi == 0) ? n[0] : (gi == 2) ? n[2] : (gi == 3) ? n[3] : n[4];
    }
    float sel[11]; long long q = nsel;
#pragma unroll
    for (int k = 0; k < K_DIG; ++k) { sel[k] = valid * (float)(q % 10); q /= 10; }
    sel[10] = valid * sgn;
    int highest = -1;
#pragma unroll
    for (int k = 0; k < K_DIG; ++k) if (sel[k] != 0.f) highest = k;
    int nsig = (highest >= 0) ? highest + 1 : 1;
#pragma unroll
    for (int k = 0; k < K_DIG; ++k) {
      int src = nsig - 1 - k; src = src < 0 ? 0 : (src > 9 ? 9 : src);
      msbS[m][k] = (k < nsig) ? sel[src] : -1.f;
    }
    msbS[m][10] = sel[10];
  }
  __syncthreads();

  // ---- stage 6: h' = h + sigmoid(gate)*has_eq*(msb @ Winj + binj) ----
  {
    float gl  = gate[0];
    int   heq = meta[2 * brow + 1];
    float geq = (1.f / (1.f + __expf(-gl))) * (heq ? 1.f : 0.f);
    for (int c = tid; c < D_MODEL / 4; c += 128) {
      float4 wv[11];
#pragma unroll
      for (int j = 0; j < 11; ++j) wv[j] = *(const float4*)(Winj + j * D_MODEL + c * 4);
      float4 bj = *(const float4*)(binj + c * 4);
      int cl = lcol(c * 4);
#pragma unroll
      for (int m = 0; m < TILE_M; ++m) {
        float4 s = bj;
#pragma unroll
        for (int j = 0; j < 11; ++j) {
          float wm = msbS[m][j];
          s.x += wm * wv[j].x; s.y += wm * wv[j].y;
          s.z += wm * wv[j].z; s.w += wm * wv[j].w;
        }
        float4 hv = *(const float4*)(hS + m * HS_STRIDE + cl);
        float4 o;
        o.x = hv.x + geq * s.x; o.y = hv.y + geq * s.y;
        o.z = hv.z + geq * s.z; o.w = hv.w + geq * s.w;
        *(float4*)(out + (size_t)(row0 + m) * D_MODEL + c * 4) = o;
      }
    }
  }
}

extern "C" void kernel_launch(void* const* d_in, const int* in_sizes, int n_in,
                              void* d_out, int out_size, void* d_ws, size_t ws_size,
                              hipStream_t stream) {
  (void)in_sizes; (void)n_in; (void)out_size; (void)ws_size;
  const float* h    = (const float*)d_in[0];
  const float* W1   = (const float*)d_in[1];
  const float* b1   = (const float*)d_in[2];
  const float* W2   = (const float*)d_in[3];
  const float* b2   = (const float*)d_in[4];
  const float* Winj = (const float*)d_in[5];
  const float* binj = (const float*)d_in[6];
  const float* gate = (const float*)d_in[7];
  const int*   ids  = (const int*)d_in[8];
  float* out = (float*)d_out;

  _Float16* W1T = (_Float16*)d_ws;                               // 128*768*2  = 196608 B
  _Float16* W2T = (_Float16*)((char*)d_ws + 196608);             // 208*128*2  =  53248 B
  int*      meta = (int*)((char*)d_ws + 196608 + 53248);         // 16*2 ints

  prep_weights<<<384, 256, 0, stream>>>(W1, W2, W1T, W2T);
  prep_meta<<<16, 256, 0, stream>>>(ids, meta);
  fused_tile<<<NBLK, 128, 0, stream>>>(h, W1T, b1, W2T, b2, Winj, binj, gate, meta, out);
}
